// MultiHeadAttention_36429912605359
// MI455X (gfx1250) — compile-verified, matrix operand order checked by reading, not run on hardware
//
#include <hip/hip_runtime.h>
#include <hip/hip_bf16.h>
#include <math.h>

// ---------------------------------------------------------------------------
// MultiHeadAttention for MI455X (gfx1250): bf16 WMMA everywhere.
//   qkv = x @ w_qkv ; flash-attention per head ; out = attn @ w_o
// Data movement: GLOBAL_LOAD_ASYNC_TO_LDS_B128 (ASYNCcnt) where no transpose
// is needed; manual transposing stores otherwise.
// ---------------------------------------------------------------------------

typedef __attribute__((ext_vector_type(16))) __bf16 v16bf;
typedef __attribute__((ext_vector_type(8)))  float  v8f;
typedef __attribute__((__vector_size__(16))) int    vi4;
typedef unsigned int  u32;
typedef unsigned short u16;

union FragU { u32 u[8]; v16bf v; };   // 16 bf16 per lane = one A/B fragment
union AccU  { v8f v; float f[8]; };   // 16x16 f32 C/D fragment per lane

#define NHEADS   16
#define HDIM     64
#define DMODEL   1024
#define SEQ      2048
#define BATCH    4
#define MTOT     (BATCH*SEQ)          // 8192 rows

#if __has_builtin(__builtin_amdgcn_global_load_async_to_lds_b128)
#define HAVE_ASYNC_LDS 1
#else
#define HAVE_ASYNC_LDS 0
#endif

typedef __attribute__((address_space(1))) vi4 gvi4;
typedef __attribute__((address_space(3))) vi4 lvi4;

// Async global->LDS 16B copy (per-lane addresses). Falls back to a
// synchronous VGPR round-trip when the builtin is unavailable.
__device__ __forceinline__ void copy_b128_to_lds(u16* lds_p, const u16* g_p) {
#if HAVE_ASYNC_LDS
  __builtin_amdgcn_global_load_async_to_lds_b128(
      (gvi4*)g_p, (lvi4*)lds_p, 0, 0);
#else
  *(uint4*)lds_p = *(const uint4*)g_p;
#endif
}

// Wait for all outstanding async global->LDS copies from this wave.
__device__ __forceinline__ void wait_async_copies() {
#if HAVE_ASYNC_LDS
#if __has_builtin(__builtin_amdgcn_s_wait_asynccnt)
  __builtin_amdgcn_s_wait_asynccnt(0);
#else
  asm volatile("s_wait_asynccnt 0x0" ::: "memory");
#endif
#endif
}

// fp32 -> bf16 round-to-nearest-even
__device__ __forceinline__ u16 f32_bf16(float f) {
  u32 x = __float_as_uint(f);
  x += 0x7FFFu + ((x >> 16) & 1u);
  return (u16)(x >> 16);
}

__device__ __forceinline__ v8f v8f_zero() {
  v8f z;
#pragma unroll
  for (int i = 0; i < 8; ++i) z[i] = 0.0f;
  return z;
}

// Load one 16-bit 16x32 A-fragment (or 32x16 B-fragment, which uses the same
// per-lane striping with "row" = B column) from a row-major [16][stride]
// bf16 tile. Per ISA 7.12.2: lane L(0-15) holds row L, K=0..7 & 16..23;
// lane L+16 holds row L, K=8..15 & 24..31 -> per (half,group) the 8 bf16 are
// contiguous, so each fragment = two 16B loads.
__device__ __forceinline__ void load_frag(const u16* base, int stride, FragU& f) {
  const int lane = threadIdx.x & 31;
  const u16* p = base + (lane & 15) * stride + (lane >> 4) * 8;
  uint4 lo = *(const uint4*)(p);
  uint4 hi = *(const uint4*)(p + 16);
  f.u[0] = lo.x; f.u[1] = lo.y; f.u[2] = lo.z; f.u[3] = lo.w;
  f.u[4] = hi.x; f.u[5] = hi.y; f.u[6] = hi.z; f.u[7] = hi.w;
}

// ---------------------------------------------------------------------------
// fp32 -> bf16 conversion (element counts are all multiples of 4)
// ---------------------------------------------------------------------------
__global__ void __launch_bounds__(256) cvt_bf16_kernel(const float* __restrict__ in,
                                                       u16* __restrict__ out, int n4) {
  int i = blockIdx.x * blockDim.x + threadIdx.x;
  int stride = gridDim.x * blockDim.x;
  for (; i < n4; i += stride) {
    float4 v = ((const float4*)in)[i];
    u32 lo = (u32)f32_bf16(v.x) | ((u32)f32_bf16(v.y) << 16);
    u32 hi = (u32)f32_bf16(v.z) | ((u32)f32_bf16(v.w) << 16);
    ((uint2*)out)[i] = make_uint2(lo, hi);
  }
}

// ---------------------------------------------------------------------------
// GEMM core: C[128x128 tile] = A[M,K] * B[K,N], bf16 in, f32 acc.
// 256 threads = 8 waves in a 2(M) x 4(N) grid; each wave 64x32 = 4x2 WMMA tiles.
// ---------------------------------------------------------------------------
#define BM 128
#define BN 128
#define BK 32
#define LDT 40   // padded LDS row stride in bf16 elems (80B, 16B aligned)

__device__ __forceinline__ void gemm_core(const u16* __restrict__ A,
                                          const u16* __restrict__ Bw,
                                          int K, int N,
                                          v8f acc[4][2],
                                          u16* As, u16* Bs) {
  const int t = threadIdx.x;
  const int wave = t >> 5;
  const int waveM = wave >> 2;   // 0..1
  const int waveN = wave & 3;    // 0..3
  const int bM = blockIdx.y * BM;
  const int bN = blockIdx.x * BN;

#pragma unroll
  for (int mt = 0; mt < 4; ++mt)
#pragma unroll
    for (int nt = 0; nt < 2; ++nt) acc[mt][nt] = v8f_zero();

  for (int k0 = 0; k0 < K; k0 += BK) {
    // Stage A tile [128][32] row-major (pairs along K contiguous) via the
    // async global->LDS engine: no VGPR data movement, tracked by ASYNCcnt.
#pragma unroll
    for (int s2 = 0; s2 < 2; ++s2) {
      int slot = t + s2 * 256;          // 512 slots: 128 rows x 4 groups of 8
      int row = slot >> 2;
      int cg  = slot & 3;
      copy_b128_to_lds(As + row * LDT + cg * 8,
                       A + (size_t)(bM + row) * K + k0 + cg * 8);
    }
    // Stage B tile transposed: Bs[n][k] so pairs along K are contiguous.
#pragma unroll
    for (int s2 = 0; s2 < 2; ++s2) {
      int slot = t + s2 * 256;          // 512 slots: 32 k-rows x 16 groups of 8
      int kr = slot >> 4;
      int ng = slot & 15;
      const u16* src = Bw + (size_t)(k0 + kr) * N + bN + ng * 8;
      uint4 d = *(const uint4*)src;
      if (k0 + BK < K) __builtin_prefetch(src + (size_t)BK * N, 0, 1);
      const u16* e = (const u16*)&d;
#pragma unroll
      for (int i = 0; i < 8; ++i)
        Bs[(ng * 8 + i) * LDT + kr] = e[i];
    }
    wait_async_copies();
    __syncthreads();

    FragU bf[2];
#pragma unroll
    for (int nt = 0; nt < 2; ++nt)
      load_frag(Bs + (waveN * 32 + nt * 16) * LDT, LDT, bf[nt]);
#pragma unroll
    for (int mt = 0; mt < 4; ++mt) {
      FragU af;
      load_frag(As + (waveM * 64 + mt * 16) * LDT, LDT, af);
#pragma unroll
      for (int nt = 0; nt < 2; ++nt)
        acc[mt][nt] = __builtin_amdgcn_wmma_f32_16x16x32_bf16(
            false, af.v, false, bf[nt].v, (short)0, acc[mt][nt], false, false);
    }
    __syncthreads();
  }
}

// QKV projection: scatter epilogue into Q/K/V [B,H,S,Dh] bf16.
__global__ void __launch_bounds__(256) gemm_qkv_kernel(const u16* __restrict__ A,
                                                       const u16* __restrict__ Bw,
                                                       u16* __restrict__ Q,
                                                       u16* __restrict__ Kd,
                                                       u16* __restrict__ V) {
  __shared__ u16 As[BM * LDT];
  __shared__ u16 Bs[BN * LDT];
  v8f acc[4][2];
  gemm_core(A, Bw, DMODEL, 3 * DMODEL, acc, As, Bs);

  const int lane = threadIdx.x & 31;
  const int wave = threadIdx.x >> 5;
  const int nl = lane & 15;
  const int mb = (lane >> 4) * 8;
  const int m0 = blockIdx.y * BM + (wave >> 2) * 64;
  const int n0 = blockIdx.x * BN + (wave & 3) * 32;
#pragma unroll
  for (int mt = 0; mt < 4; ++mt)
#pragma unroll
    for (int nt = 0; nt < 2; ++nt) {
      AccU a; a.v = acc[mt][nt];
#pragma unroll
      for (int r = 0; r < 8; ++r) {
        int m = m0 + mt * 16 + mb + r;          // row in [0,8192)
        int n = n0 + nt * 16 + nl;              // col in [0,3072)
        int which = n >> 10, rem = n & 1023;
        int h = rem >> 6, d = rem & 63;
        int bb = m >> 11, s = m & 2047;
        size_t idx = (((size_t)(bb * NHEADS + h) * SEQ + s) * HDIM + d);
        u16 val = f32_bf16(a.f[r]);
        (which == 0 ? Q : which == 1 ? Kd : V)[idx] = val;
      }
    }
}

// Output projection: out = attnO @ w_o, fp32 epilogue straight to d_out.
__global__ void __launch_bounds__(256) gemm_out_kernel(const u16* __restrict__ A,
                                                       const u16* __restrict__ Bw,
                                                       float* __restrict__ C) {
  __shared__ u16 As[BM * LDT];
  __shared__ u16 Bs[BN * LDT];
  v8f acc[4][2];
  gemm_core(A, Bw, DMODEL, DMODEL, acc, As, Bs);

  const int lane = threadIdx.x & 31;
  const int wave = threadIdx.x >> 5;
  const int nl = lane & 15;
  const int mb = (lane >> 4) * 8;
  const int m0 = blockIdx.y * BM + (wave >> 2) * 64;
  const int n0 = blockIdx.x * BN + (wave & 3) * 32;
#pragma unroll
  for (int mt = 0; mt < 4; ++mt)
#pragma unroll
    for (int nt = 0; nt < 2; ++nt) {
      AccU a; a.v = acc[mt][nt];
#pragma unroll
      for (int r = 0; r < 8; ++r) {
        int m = m0 + mt * 16 + mb + r;
        int n = n0 + nt * 16 + nl;
        C[(size_t)m * DMODEL + n] = a.f[r];
      }
    }
}

// ---------------------------------------------------------------------------
// Flash attention: block = one (b, h, 128 q-rows); wave owns 16 q-rows.
// Loop over 32-key chunks: S = Q K^T (4 WMMA), online softmax, O += P V (4 WMMA).
// ---------------------------------------------------------------------------
#define KSTR 72   // Ks row stride (bf16): 144B, 16B aligned
#define VSTR 40   // Vt row stride (bf16): 80B, 16B aligned

__global__ void __launch_bounds__(256) attn_kernel(const u16* __restrict__ Q,
                                                   const u16* __restrict__ Kd,
                                                   const u16* __restrict__ V,
                                                   u16* __restrict__ O) {
  __shared__ u16 Ks[32 * KSTR];        // K chunk  [key][d]   (pairs along d)
  __shared__ u16 Vt[64 * VSTR];        // V chunk  [d][key]   (pairs along key)
  __shared__ u16 Ps[8 * 16 * 32];      // per-wave P scratch [16][32]

  const int t = threadIdx.x;
  const int lane = t & 31;
  const int wave = t >> 5;
  const int b = blockIdx.z, h = blockIdx.y;
  const int q0 = blockIdx.x * 128 + wave * 16;
  const size_t bh = (size_t)(b * NHEADS + h) * SEQ;
  const float scale = 0.125f;          // 1/sqrt(64 + 1e-9)

  // Q fragments for this wave's 16 rows (d-split 0..31 / 32..63), kept in VGPRs.
  FragU qf[2];
  load_frag(Q + (bh + q0) * HDIM + 0,  HDIM, qf[0]);
  load_frag(Q + (bh + q0) * HDIM + 32, HDIM, qf[1]);

  AccU o[4];
  float rmax[8], rsum[8];
#pragma unroll
  for (int i = 0; i < 4; ++i) o[i].v = v8f_zero();
#pragma unroll
  for (int r = 0; r < 8; ++r) { rmax[r] = -1e30f; rsum[r] = 0.0f; }

  const int key_t = t >> 3, dg = t & 7;   // staging: 256 slots = 32 keys x 8 dgroups

  for (int kc = 0; kc < SEQ / 32; ++kc) {
    // ---- stage K chunk (async, as-is) and V chunk (manual transpose) ----
    {
      size_t src = (bh + (size_t)kc * 32 + key_t) * HDIM + dg * 8;
      copy_b128_to_lds(Ks + key_t * KSTR + dg * 8, Kd + src);
      uint4 dv = *(const uint4*)(V + src);
      const u16* e = (const u16*)&dv;
#pragma unroll
      for (int i = 0; i < 8; ++i)
        Vt[(dg * 8 + i) * VSTR + key_t] = e[i];
    }
    wait_async_copies();
    __syncthreads();

    // ---- scores: S[16 q x 32 k] = Q (16x64) * K^T (64x32) ----
    AccU sc[2];
#pragma unroll
    for (int kt = 0; kt < 2; ++kt) {
      v8f s = v8f_zero();
#pragma unroll
      for (int ds = 0; ds < 2; ++ds) {
        FragU kb;
        load_frag(Ks + (kt * 16) * KSTR + ds * 32, KSTR, kb);
        s = __builtin_amdgcn_wmma_f32_16x16x32_bf16(
            false, qf[ds].v, false, kb.v, (short)0, s, false, false);
      }
      sc[kt].v = s;
    }

    // ---- online softmax (row stats live replicated in each half-wave) ----
    float corr[8];
#pragma unroll
    for (int r = 0; r < 8; ++r) {
      float s0 = sc[0].f[r] * scale;
      float s1 = sc[1].f[r] * scale;
      float a = fmaxf(s0, s1);
#pragma unroll
      for (int msk = 1; msk < 16; msk <<= 1)
        a = fmaxf(a, __shfl_xor(a, msk, 32));
      float mn = fmaxf(rmax[r], a);
      corr[r] = __expf(rmax[r] - mn);
      rmax[r] = mn;
      float p0 = __expf(s0 - mn);
      float p1 = __expf(s1 - mn);
      sc[0].f[r] = p0; sc[1].f[r] = p1;
      float ls = p0 + p1;
#pragma unroll
      for (int msk = 1; msk < 16; msk <<= 1)
        ls += __shfl_xor(ls, msk, 32);
      rsum[r] = rsum[r] * corr[r] + ls;
    }
#pragma unroll
    for (int dt = 0; dt < 4; ++dt)
#pragma unroll
      for (int r = 0; r < 8; ++r) o[dt].f[r] *= corr[r];

    // ---- C-layout -> A-layout for P via wave-private LDS (bf16) ----
    u16* Pw = Ps + wave * 512;
    const int nl = lane & 15;
    const int mb8 = (lane >> 4) * 8;
#pragma unroll
    for (int kt = 0; kt < 2; ++kt)
#pragma unroll
      for (int r = 0; r < 8; ++r)
        Pw[(mb8 + r) * 32 + kt * 16 + nl] = f32_bf16(sc[kt].f[r]);
    // LDS is in-order per wave; the wait + memory clobber also stops the
    // compiler from hoisting the fragment loads above the stores.
    asm volatile("s_wait_dscnt 0x0" ::: "memory");

    FragU pf;
    load_frag(Pw, 32, pf);

    // ---- O[16x64] += P (16x32) * V (32x64) ----
#pragma unroll
    for (int dt = 0; dt < 4; ++dt) {
      FragU vb;
      load_frag(Vt + (dt * 16) * VSTR, VSTR, vb);
      o[dt].v = __builtin_amdgcn_wmma_f32_16x16x32_bf16(
          false, pf.v, false, vb.v, (short)0, o[dt].v, false, false);
    }
    __syncthreads();   // protect Ks/Vt before next chunk's staging
  }

  // ---- normalize and store O as [B,S,H*Dh] bf16 ----
  const int nl = lane & 15;
  const int mb8 = (lane >> 4) * 8;
#pragma unroll
  for (int r = 0; r < 8; ++r) {
    float inv = 1.0f / rsum[r];
    int qrow = q0 + mb8 + r;
    size_t rowbase = ((size_t)b * SEQ + qrow) * DMODEL + h * HDIM;
#pragma unroll
    for (int dt = 0; dt < 4; ++dt)
      O[rowbase + dt * 16 + nl] = f32_bf16(o[dt].f[r] * inv);
  }
}

// ---------------------------------------------------------------------------
// Launch: cvt x3 -> QKV GEMM -> flash attention -> output GEMM
// ---------------------------------------------------------------------------
extern "C" void kernel_launch(void* const* d_in, const int* in_sizes, int n_in,
                              void* d_out, int out_size, void* d_ws, size_t ws_size,
                              hipStream_t stream) {
  const float* x      = (const float*)d_in[0];
  const float* w_qkv  = (const float*)d_in[1];
  const float* w_o    = (const float*)d_in[2];
  float* out = (float*)d_out;

  char* ws = (char*)d_ws;
  size_t off = 0;
  u16* xb    = (u16*)(ws + off); off += (size_t)MTOT * DMODEL * 2;        // 16.8 MB
  u16* wqkvb = (u16*)(ws + off); off += (size_t)DMODEL * 3 * DMODEL * 2;  //  6.3 MB
  u16* wob   = (u16*)(ws + off); off += (size_t)DMODEL * DMODEL * 2;      //  2.1 MB
  u16* Qb    = (u16*)(ws + off); off += (size_t)MTOT * DMODEL * 2;        // 16.8 MB
  u16* Kb    = (u16*)(ws + off); off += (size_t)MTOT * DMODEL * 2;
  u16* Vb    = (u16*)(ws + off); off += (size_t)MTOT * DMODEL * 2;
  u16* Ob    = (u16*)(ws + off); off += (size_t)MTOT * DMODEL * 2;        // ~92 MB total

  int n_x = MTOT * DMODEL / 4;
  int n_wq = DMODEL * 3 * DMODEL / 4;
  int n_wo = DMODEL * DMODEL / 4;
  cvt_bf16_kernel<<<dim3(2048), dim3(256), 0, stream>>>(x, xb, n_x);
  cvt_bf16_kernel<<<dim3(1024), dim3(256), 0, stream>>>(w_qkv, wqkvb, n_wq);
  cvt_bf16_kernel<<<dim3(512),  dim3(256), 0, stream>>>(w_o, wob, n_wo);

  // qkv = x @ w_qkv : M=8192, N=3072, K=1024
  gemm_qkv_kernel<<<dim3(3 * DMODEL / BN, MTOT / BM), dim3(256), 0, stream>>>(
      xb, wqkvb, Qb, Kb, Vb);

  // flash attention: grid = (S/128, H, B)
  attn_kernel<<<dim3(SEQ / 128, NHEADS, BATCH), dim3(256), 0, stream>>>(
      Qb, Kb, Vb, Ob);

  // out = attnO @ w_o : M=8192, N=1024, K=1024
  gemm_out_kernel<<<dim3(DMODEL / BN, MTOT / BM), dim3(256), 0, stream>>>(
      Ob, wob, out);
}
